// TorchAttn_4363686772831
// MI455X (gfx1250) — matrix operand order repaired and checked by reading, not compiled, not run on hardware
//
#include <hip/hip_runtime.h>

typedef __bf16 bf16;
typedef bf16  v16bf __attribute__((ext_vector_type(16)));
typedef bf16  v8bf  __attribute__((ext_vector_type(8)));
typedef float v8f   __attribute__((ext_vector_type(8)));

#define BB 32
#define HH 8
#define DD 256
#define DH 32
#define NN 504
#define NP 512
#define TT 12160
#define M1 (BB*NN)          // 16128 rows, 1008 tiles of 16

// ---------------- WMMA fragment helpers (CDNA5 ISA 7.12.2 layouts) ----------

// A fragment 16x32 bf16, row-major source, row stride ld (elements).
// lane L holds row m0 + L%16; element e -> K = (e/8)*16 + (L/16)*8 + e%8
__device__ __forceinline__ v16bf load_a_frag(const bf16* base, int m0, int ld, int k0) {
  int lane = threadIdx.x & 31;
  int lg   = lane >> 4;
  int r    = lane & 15;
  const bf16* rowp = base + (size_t)(m0 + r) * ld + k0;
  v8bf lo = *(const v8bf*)(rowp + lg * 8);        // K = lg*8 .. lg*8+7
  v8bf hi = *(const v8bf*)(rowp + 16 + lg * 8);   // K = 16+lg*8 ..
  v16bf a;
#pragma unroll
  for (int i = 0; i < 8; i++) { a[i] = lo[i]; a[8 + i] = hi[i]; }
  return a;
}

// B fragment 32x16 bf16 where source is row-major [Ncols][K] (i.e. B[k][n] = src[n][k]).
// lane L holds column n0 + L%16; element e -> K = (e/8)*16 + (L/16)*8 + e%8
__device__ __forceinline__ v16bf load_b_frag(const bf16* base, int n0, int ld, int k0) {
  int lane = threadIdx.x & 31;
  int lg   = lane >> 4;
  int c    = lane & 15;
  const bf16* rowp = base + (size_t)(n0 + c) * ld + k0;
  v8bf lo = *(const v8bf*)(rowp + lg * 8);
  v8bf hi = *(const v8bf*)(rowp + 16 + lg * 8);
  v16bf b;
#pragma unroll
  for (int i = 0; i < 8; i++) { b[i] = lo[i]; b[8 + i] = hi[i]; }
  return b;
}

__device__ __forceinline__ v8f wmma_bf16(v16bf a, v16bf b, v8f c) {
  return __builtin_amdgcn_wmma_f32_16x16x32_bf16(false, a, false, b, (short)0, c, false, false);
}

// ---------------- Kernel 0: zero dense h ------------------------------------
__global__ void k_zero_h(bf16* h) {
  size_t i = (size_t)blockIdx.x * 256 + threadIdx.x;
  h[i] = (bf16)0.0f;
}

// ---------------- Kernel 1: to_dense_batch scatter (f32 -> bf16) ------------
__global__ void k_scatter(const float* __restrict__ x, const int* __restrict__ gid,
                          const int* __restrict__ pos, bf16* __restrict__ h) {
  size_t i = (size_t)blockIdx.x * 256 + threadIdx.x;   // < TT*DD
  int t = (int)(i >> 8);
  int d = (int)(i & 255);
  int g = gid[t], p = pos[t];
  h[((size_t)(g * NN + p) << 8) + d] = (bf16)__builtin_nontemporal_load(&x[i]);
}

// ---------------- Kernel 2: weight conversion f32 -> bf16 -------------------
__global__ void k_convw(const float* __restrict__ wqkv, const float* __restrict__ wout,
                        bf16* __restrict__ wq, bf16* __restrict__ wo) {
  int i = blockIdx.x * 256 + threadIdx.x;
  if (i < 768 * 256) wq[i] = (bf16)wqkv[i];
  else               wo[i - 768 * 256] = (bf16)wout[i - 768 * 256];
}

// ---------------- Kernel 3: fused QKV projection GEMM -----------------------
// C[16128, 768] = h[16128,256] @ W^T (W row-major [768,256]); scatter into
// q [BH,NP,32] (pre-scaled 1/sqrt(dh)), k [BH,NP,32], vT [BH,32,NP] as bf16.
__global__ void __launch_bounds__(256) k_qkv(const bf16* __restrict__ h,
                                             const bf16* __restrict__ w,
                                             const float* __restrict__ bias,
                                             bf16* __restrict__ q,
                                             bf16* __restrict__ k,
                                             bf16* __restrict__ vT) {
  int wave  = (blockIdx.x * 256 + threadIdx.x) >> 5;   // 48384 waves
  int mt    = wave / 48;
  int nt    = wave % 48;
  int m0    = mt * 16;
  int ncol0 = nt * 16;
  v8f acc = {};
#pragma unroll
  for (int k0 = 0; k0 < DD; k0 += 32) {
    v16bf a = load_a_frag(h, m0, DD, k0);
    v16bf b = load_b_frag(w, ncol0, DD, k0);
    acc = wmma_bf16(a, b, acc);
  }
  int lane = threadIdx.x & 31, lg = lane >> 4, cc = lane & 15;
  int col  = ncol0 + cc;
  int sect = col >> 8;          // 0=q 1=k 2=v
  int cs   = col & 255;
  int head = cs >> 5;
  int d    = cs & 31;
  float bv = bias[col];
#pragma unroll
  for (int j = 0; j < 8; j++) {
    int m  = m0 + j + 8 * lg;
    int b_ = m / NN;
    int n  = m % NN;
    float val = acc[j] + bv;
    size_t hb = (size_t)(b_ * HH + head);
    if (sect == 0)      q [(hb * NP + n) * DH + d] = (bf16)(val * 0.1767766952966369f);
    else if (sect == 1) k [(hb * NP + n) * DH + d] = (bf16)val;
    else                vT[(hb * DH + d) * NP + n] = (bf16)val;
  }
}

// ---------------- Kernel 4: attention (scores + mask + softmax + AV) --------
// one block = one (b, h, 16-row q tile); 128 threads = 4 waves.
// attn_bias (260 MB, streamed once, > 192 MB L2) and adj (8 MB) use NT loads.
__global__ void __launch_bounds__(128) k_attn(const bf16* __restrict__ q,
                                              const bf16* __restrict__ kk,
                                              const bf16* __restrict__ vT,
                                              const float* __restrict__ bias,
                                              const unsigned char* __restrict__ adj,
                                              const unsigned char* __restrict__ mask,
                                              bf16* __restrict__ ao) {
  __shared__ __align__(16) float S[16 * 520];
  __shared__ __align__(16) bf16  P[16 * 520];

  int blk = blockIdx.x;               // b*H*32 + h*32 + qt
  int qt  = blk & 31;
  int bh  = blk >> 5;
  int hh_ = bh & 7;
  int b   = bh >> 3;
  int q0  = qt * 16;
  int wave = threadIdx.x >> 5;
  int lane = threadIdx.x & 31, lg = lane >> 4, cc = lane & 15;

  const bf16* qbase = q  + (size_t)bh * NP * DH;
  const bf16* kbase = kk + (size_t)bh * NP * DH;
  const bf16* vbase = vT + (size_t)bh * DH * NP;
  const float* bbase = bias + (size_t)bh * NN * NN;
  const unsigned char* abase = adj + (size_t)b * NN * NN;
  const unsigned char* mbase = mask + (size_t)b * NN;

  // q tile A fragment: dh == 32 == WMMA K, loaded once
  v16bf aq = load_a_frag(qbase, q0, DH, 0);

  // score tiles: 32 key-tiles of 16, round-robin over 4 waves
  for (int kt = wave; kt < 32; kt += 4) {
    int key0 = kt * 16;
    v16bf bk = load_b_frag(kbase, key0, DH, 0);
    v8f s = {};
    s = wmma_bf16(aq, bk, s);
    int kcol = key0 + cc;
    bool kvalid = (kcol < NN) && (mbase[kcol] != 0);
#pragma unroll
    for (int j = 0; j < 8; j++) {
      int qi = q0 + j + 8 * lg;
      float val = -1e9f;
      if (kvalid && qi < NN) {
        bool allow = (__builtin_nontemporal_load(&abase[(size_t)qi * NN + kcol]) != 0)
                     || (qi == kcol);
        if (allow) val = s[j] + __builtin_nontemporal_load(&bbase[(size_t)qi * NN + kcol]);
      }
      S[(j + 8 * lg) * 520 + kcol] = val;
    }
  }
  __syncthreads();

  // row softmax: row r handled by 8 consecutive lanes, 64 cols each
  {
    int r = threadIdx.x >> 3, seg = threadIdx.x & 7;
    float* row = S + r * 520 + seg * 64;
    float mx = -3.0e38f;
    for (int c = 0; c < 64; c++) mx = fmaxf(mx, row[c]);
    for (int off = 1; off < 8; off <<= 1) mx = fmaxf(mx, __shfl_xor(mx, off, 32));
    float sum = 0.f;
    for (int c = 0; c < 64; c++) { float e = __expf(row[c] - mx); row[c] = e; sum += e; }
    for (int off = 1; off < 8; off <<= 1) sum += __shfl_xor(sum, off, 32);
    float inv = 1.0f / sum;
    bf16* prow = P + r * 520 + seg * 64;
    for (int c = 0; c < 64; c++) prow[c] = (bf16)(row[c] * inv);
  }
  __syncthreads();

  // AV: out[16, 32] = P[16,512] @ V[512,32]; waves 0/1 take d0 = 0/16
  if (wave < 2) {
    int d0 = wave * 16;
    v8f acc = {};
#pragma unroll
    for (int k0 = 0; k0 < NP; k0 += 32) {
      v16bf ap = load_a_frag(P, 0, 520, k0);
      v16bf bv = load_b_frag(vbase, d0, NP, k0);
      acc = wmma_bf16(ap, bv, acc);
    }
#pragma unroll
    for (int j = 0; j < 8; j++) {
      int qi = q0 + j + 8 * lg;     // < NP (pad rows dropped at gather)
      ao[(((size_t)b * NP + qi) << 8) + hh_ * 32 + d0 + cc] = (bf16)acc[j];
    }
  }
}

// ---------------- Kernel 5: out-proj GEMM with fused ragged gather ----------
// out[T,256] = ao[gather rows][256] @ out_w^T + out_b ;  T = 12160 = 760 tiles
__global__ void __launch_bounds__(256) k_out(const bf16* __restrict__ ao,
                                             const bf16* __restrict__ wo,
                                             const float* __restrict__ ob,
                                             const int* __restrict__ gid,
                                             const int* __restrict__ pos,
                                             float* __restrict__ out) {
  int wave = (blockIdx.x * 256 + threadIdx.x) >> 5;  // 12160 waves
  int mt = wave >> 4;            // 760 row tiles
  int nt = wave & 15;            // 16 col tiles
  int t0 = mt * 16, n0 = nt * 16;
  int lane = threadIdx.x & 31, lg = lane >> 4, cc = lane & 15;

  int tr = t0 + cc;                                   // gathered A row per lane
  const bf16* rowp = ao + (((size_t)gid[tr] * NP + pos[tr]) << 8);

  v8f acc = {};
#pragma unroll
  for (int k0 = 0; k0 < DD; k0 += 32) {
    v8bf lo = *(const v8bf*)(rowp + k0 + lg * 8);
    v8bf hi = *(const v8bf*)(rowp + k0 + 16 + lg * 8);
    v16bf a;
#pragma unroll
    for (int i = 0; i < 8; i++) { a[i] = lo[i]; a[8 + i] = hi[i]; }
    v16bf bfr = load_b_frag(wo, n0, DD, k0);
    acc = wmma_bf16(a, bfr, acc);
  }
  float bv = ob[n0 + cc];
#pragma unroll
  for (int j = 0; j < 8; j++) {
    int t = t0 + j + 8 * lg;
    __builtin_nontemporal_store(acc[j] + bv, &out[(size_t)t * DD + n0 + cc]);
  }
}

// ---------------- host launcher ---------------------------------------------
extern "C" void kernel_launch(void* const* d_in, const int* in_sizes, int n_in,
                              void* d_out, int out_size, void* d_ws, size_t ws_size,
                              hipStream_t stream) {
  const float* x          = (const float*)d_in[0];
  const float* attn_bias  = (const float*)d_in[1];
  const float* in_proj_w  = (const float*)d_in[2];
  const float* in_proj_b  = (const float*)d_in[3];
  const float* out_w      = (const float*)d_in[4];
  const float* out_b      = (const float*)d_in[5];
  const unsigned char* adj  = (const unsigned char*)d_in[6];
  const unsigned char* mask = (const unsigned char*)d_in[7];
  const int* gid          = (const int*)d_in[8];
  const int* pos          = (const int*)d_in[9];
  float* out              = (float*)d_out;

  char* ws = (char*)d_ws;
  size_t off = 0;
  auto carve = [&](size_t bytes) {
    char* p = ws + off;
    off += (bytes + 255) & ~(size_t)255;
    return p;
  };
  bf16* h  = (bf16*)carve((size_t)M1 * DD * 2);             // 8.26 MB
  bf16* wq = (bf16*)carve((size_t)768 * 256 * 2);
  bf16* wo = (bf16*)carve((size_t)256 * 256 * 2);
  bf16* qb = (bf16*)carve((size_t)BB * HH * NP * DH * 2);   // 8.39 MB
  bf16* kb = (bf16*)carve((size_t)BB * HH * NP * DH * 2);
  bf16* vb = (bf16*)carve((size_t)BB * HH * DH * NP * 2);
  bf16* ao = (bf16*)carve((size_t)BB * NP * DD * 2);        // 8.39 MB

  k_zero_h <<<M1 * DD / 256, 256, 0, stream>>>(h);
  k_scatter<<<TT,             256, 0, stream>>>(x, gid, pos, h);
  k_convw  <<<1024,           256, 0, stream>>>(in_proj_w, out_w, wq, wo);
  k_qkv    <<<6048,           256, 0, stream>>>(h, wq, in_proj_b, qb, kb, vb);
  k_attn   <<<BB * HH * 32,   128, 0, stream>>>(qb, kb, vb, attn_bias, adj, mask, ao);
  k_out    <<<1520,           256, 0, stream>>>(ao, wo, out_b, gid, pos, out);
}
